// TFDecepticon_71811853189671
// MI455X (gfx1250) — compile-verified, matrix operand order checked by reading, not run on hardware
//
#include <hip/hip_runtime.h>
#include <math.h>
#include <stdint.h>

// ---------------- problem constants ----------------
#define BB    8      // batch
#define CC    64     // channels
#define TMAX  1000   // time
#define TP    1024   // padded time (multiple of 32)
#define FF    64     // freq
#define HH    4      // heads
#define HS    4      // q/k hidden per head
#define CV    16     // v channels per head
#define HBN   32     // H*B
#define DQK   256    // HS*F
#define DV    1024   // CV*F
#define EPS   1e-5f
#define NKT   (TP / 32)   // key-tile steps

typedef __attribute__((ext_vector_type(16))) _Float16 v16h;
typedef __attribute__((ext_vector_type(8)))  _Float16 v8h;
typedef __attribute__((ext_vector_type(8)))  float    v8f;

// Fragment loader for WMMA f16 16x16x32 A/B operands from a row-major
// [16 rows x 32 cols] tile with row stride ldh (in halves).
// Layout (ISA 7.12.2): lanes 0-15 -> row=lane, K=0..7 & 16..23;
// lanes 16-31 -> row=lane-16, K=8..15 & 24..31. Two 16B loads per lane.
__device__ __forceinline__ v16h load_frag(const _Float16* base, int ldh) {
    int lane = threadIdx.x & 31;
    const _Float16* p = base + (lane & 15) * ldh + ((lane >> 4) << 3);
    v8h lo = *reinterpret_cast<const v8h*>(p);
    v8h hi = *reinterpret_cast<const v8h*>(p + 16);
    return __builtin_shufflevector(lo, hi, 0,1,2,3,4,5,6,7,8,9,10,11,12,13,14,15);
}

__device__ __forceinline__ v8f wmma16(v16h a, v16h b, v8f c) {
    // emits v_wmma_f32_16x16x32_f16
    return __builtin_amdgcn_wmma_f32_16x16x32_f16(false, a, false, b, (short)0, c,
                                                  false, false);
}

__device__ __forceinline__ void wait_asynccnt0() {
#if __has_builtin(__builtin_amdgcn_s_wait_asynccnt)
    __builtin_amdgcn_s_wait_asynccnt(0);
#else
    asm volatile("s_wait_asynccnt 0x0" ::: "memory");
#endif
}

// Per-lane async global->LDS 16B copy (ASYNCcnt path, no VGPR staging).
// Generic LDS pointers are (aperture_base | lds_offset); HW truncates the
// flat address to [31:0] for LDS, so the low 32 bits are the LDS address.
__device__ __forceinline__ void async_copy_b128(const v8h* g, v8h* l) {
    unsigned       ldsa = (unsigned)(uintptr_t)l;
    unsigned long long ga = (unsigned long long)(uintptr_t)g;
    asm volatile("global_load_async_to_lds_b128 %0, %1, off"
                 :: "v"(ldsa), "v"(ga) : "memory");
}

__device__ __forceinline__ int grpidx(int o) {
    return (o < 16) ? (o >> 2) : ((o < 32) ? 4 + ((o - 16) >> 2) : 8 + ((o - 32) >> 4));
}

// ============================================================
// Kernel 1: fused Q/K/V 1x1 conv + PReLU + ChanFreqNorm -> f16
//   Qh [hb][TP][256]  (scaled by 1/16 = 1/sqrt(H*F))
//   Kh [hb][TP][256]
//   Vt [hb][1024][TP] (transposed so attention A-fragments are row-major)
// one block per (t, b); t in [0,TP) — t >= TMAX writes zero padding
// ============================================================
__global__ __launch_bounds__(256) void qkv_kernel(
    const float* __restrict__ x,
    const float* __restrict__ qw, const float* __restrict__ qb,
    const float* __restrict__ qa, const float* __restrict__ qg, const float* __restrict__ qbe,
    const float* __restrict__ kw, const float* __restrict__ kb,
    const float* __restrict__ ka, const float* __restrict__ kg, const float* __restrict__ kbe,
    const float* __restrict__ vw, const float* __restrict__ vb,
    const float* __restrict__ va, const float* __restrict__ vg, const float* __restrict__ vbe,
    _Float16* __restrict__ Qh, _Float16* __restrict__ Kh, _Float16* __restrict__ Vt)
{
    const int t   = blockIdx.x;
    const int b   = blockIdx.y;
    const int tid = threadIdx.x;

    if (t >= TMAX) {  // zero the padded tail so attention reads are clean
        for (int i = tid; i < HH * DQK; i += 256) {
            int h = i >> 8, d = i & 255;
            size_t idx = ((size_t)(h * BB + b) * TP + t) * DQK + d;
            Qh[idx] = (_Float16)0.f;
            Kh[idx] = (_Float16)0.f;
        }
        for (int i = tid; i < HH * DV; i += 256) {
            int h = i >> 10, vc = i & 1023;
            Vt[((size_t)(h * BB + b) * DV + vc) * TP + t] = (_Float16)0.f;
        }
        return;
    }

    __shared__ float xs[CC * FF];     // 16 KB : x[b,:,t,:]
    __shared__ float wsm[96 * CC];    // 24 KB : [qw;kw;vw] rows
    __shared__ float gsum[12], gsq[12];

    for (int i = tid; i < CC * FF; i += 256) {
        int c = i >> 6, f = i & 63;
        xs[i] = x[(((size_t)b * CC + c) * TMAX + t) * FF + f];
    }
    for (int i = tid; i < 16 * CC; i += 256) { wsm[i] = qw[i]; wsm[16 * CC + i] = kw[i]; }
    for (int i = tid; i < 64 * CC; i += 256) wsm[32 * CC + i] = vw[i];
    if (tid < 12) { gsum[tid] = 0.f; gsq[tid] = 0.f; }
    __syncthreads();

    const int f = tid & 63;
    const int obase = tid >> 6;   // uniform within a wave
    float acc[24];

    #pragma unroll
    for (int k = 0; k < 24; k++) {
        int o = obase + 4 * k;
        const float* wr = &wsm[o * CC];
        float s = 0.f;
        for (int c = 0; c < CC; c++) s += wr[c] * xs[c * FF + f];
        float bias, alpha;
        if (o < 16)      { bias = qb[o];      alpha = qa[o >> 2]; }
        else if (o < 32) { bias = kb[o - 16]; alpha = ka[(o - 16) >> 2]; }
        else             { bias = vb[o - 32]; alpha = va[(o - 32) >> 4]; }
        s += bias;
        s = s > 0.f ? s : alpha * s;      // PReLU
        acc[k] = s;
        // wave reduce (o, hence group, is wave-uniform) then one LDS atomic
        float s1 = s, s2 = s * s;
        #pragma unroll
        for (int m = 16; m >= 1; m >>= 1) {
            s1 += __shfl_xor(s1, m, 32);
            s2 += __shfl_xor(s2, m, 32);
        }
        if ((tid & 31) == 0) {
            int g = grpidx(o);
            atomicAdd(&gsum[g], s1);
            atomicAdd(&gsq[g], s2);
        }
    }
    __syncthreads();

    #pragma unroll
    for (int k = 0; k < 24; k++) {
        int o = obase + 4 * k;
        int g = grpidx(o);
        float n   = (o < 32) ? 256.f : 1024.f;
        float mu  = gsum[g] / n;
        float var = gsq[g] / n - mu * mu;
        float istd = 1.f / (sqrtf(fmaxf(var, 0.f)) + EPS);
        float y = (acc[k] - mu) * istd;
        if (o < 16) {
            int h = o >> 2, hs = o & 3;
            y = y * qg[(h * HS + hs) * FF + f] + qbe[(h * HS + hs) * FF + f];
            Qh[((size_t)(h * BB + b) * TP + t) * DQK + hs * FF + f] =
                (_Float16)(y * 0.0625f);  // fold 1/sqrt(256)
        } else if (o < 32) {
            int oo = o - 16, h = oo >> 2, hs = oo & 3;
            y = y * kg[(h * HS + hs) * FF + f] + kbe[(h * HS + hs) * FF + f];
            Kh[((size_t)(h * BB + b) * TP + t) * DQK + hs * FF + f] = (_Float16)y;
        } else {
            int oo = o - 32, h = oo >> 4, cv = oo & 15;
            y = y * vg[(h * CV + cv) * FF + f] + vbe[(h * CV + cv) * FF + f];
            Vt[((size_t)(h * BB + b) * DV + cv * FF + f) * TP + t] = (_Float16)y;
        }
    }
}

// ============================================================
// Kernel 2: flash attention with v_wmma_f32_16x16x32_f16.
// Block = 8 waves, one (hb, 16-query tile) per block.
// Each wave: computes S^T = K.Q^T (16x16 tiles; the f32 C layout of S^T
// equals the f16 B layout of P^T after exp), runs wave-private online
// softmax (lane = query column), and applies P to its 128 V columns.
// K tiles are double-buffered in LDS via global_load_async_to_lds_b128
// (ASYNCcnt), overlapping the next tile's copy with this tile's WMMAs.
// Output written as xo [B][C][T][F] f32.
// ============================================================
__global__ __launch_bounds__(256) void attn_kernel(
    const _Float16* __restrict__ Qh, const _Float16* __restrict__ Kh,
    const _Float16* __restrict__ Vt, float* __restrict__ xo)
{
    const int qt   = blockIdx.x;           // 0..62
    const int hb   = blockIdx.y;           // 0..31
    const int wave = threadIdx.x >> 5;
    const int lane = threadIdx.x & 31;
    const int qrow0 = qt * 16;

    const _Float16* Qbase = Qh + (size_t)hb * TP * DQK + (size_t)qrow0 * DQK;
    const _Float16* Kbase = Kh + (size_t)hb * TP * DQK;
    const _Float16* Vbase = Vt + (size_t)hb * DV * TP + (size_t)wave * 128 * TP;

    // Preload Q^T B-fragments: 8 chunks of 32 along d (N = query = lane&15)
    v16h qf[8];
    #pragma unroll
    for (int d = 0; d < 8; d++) qf[d] = load_frag(Qbase + d * 32, DQK);

    v8f o_acc[8];
    const v8f vzero = {0.f,0.f,0.f,0.f,0.f,0.f,0.f,0.f};
    #pragma unroll
    for (int n = 0; n < 8; n++) o_acc[n] = vzero;

    float m_run = -1e30f, l_run = 0.f;

    // double-buffered K tile: 2 x (32 keys x 256 halves) = 32 KB
    __shared__ __align__(16) _Float16 ks_tile[2][32 * DQK];

    // prologue: async-copy tile 0 into buffer 0
    {
        const v8h* src = reinterpret_cast<const v8h*>(Kbase);
        v8h* dst = reinterpret_cast<v8h*>(&ks_tile[0][0]);
        for (int i = threadIdx.x; i < 32 * DQK / 8; i += 256)
            async_copy_b128(src + i, dst + i);
    }
    wait_asynccnt0();
    __syncthreads();

    for (int kt = 0; kt < NKT; kt++) {
        const int k0 = kt * 32;
        const int cur = kt & 1;

        // kick off async copy of the next K tile into the other buffer
        if (kt + 1 < NKT) {
            const v8h* src = reinterpret_cast<const v8h*>(Kbase + (size_t)(k0 + 32) * DQK);
            v8h* dst = reinterpret_cast<v8h*>(&ks_tile[cur ^ 1][0]);
            for (int i = threadIdx.x; i < 32 * DQK / 8; i += 256)
                async_copy_b128(src + i, dst + i);
        }

        // S^T tiles: st1 = keys k0..k0+15, st2 = keys k0+16..k0+31 (x 16 queries)
        const _Float16* kst = &ks_tile[cur][0];
        v8f st1 = vzero, st2 = vzero;
        #pragma unroll
        for (int d = 0; d < 8; d++) {
            v16h a1 = load_frag(kst + d * 32, DQK);
            v16h a2 = load_frag(kst + 16 * DQK + d * 32, DQK);
            st1 = wmma16(a1, qf[d], st1);
            st2 = wmma16(a2, qf[d], st2);
        }

        // online softmax; this lane owns query column q = lane&15
        const int keyb = k0 + ((lane >> 4) << 3);
        float p1[8], p2[8];
        float tmax = -1e30f;
        #pragma unroll
        for (int j = 0; j < 8; j++) {
            float s1 = (keyb + j      < TMAX) ? st1[j] : -1e30f;
            float s2 = (keyb + 16 + j < TMAX) ? st2[j] : -1e30f;
            p1[j] = s1; p2[j] = s2;
            tmax = fmaxf(tmax, fmaxf(s1, s2));
        }
        tmax = fmaxf(tmax, __shfl_xor(tmax, 16, 32));
        float m_new = fmaxf(m_run, tmax);
        float alpha = __expf(m_run - m_new);
        float rs = 0.f;
        #pragma unroll
        for (int j = 0; j < 8; j++) {
            p1[j] = __expf(p1[j] - m_new);
            p2[j] = __expf(p2[j] - m_new);
            rs += p1[j] + p2[j];
        }
        rs += __shfl_xor(rs, 16, 32);
        l_run = l_run * alpha + rs;
        m_run = m_new;

        // S^T C-layout == P^T B-layout: pack directly, no shuffles
        v16h pf;
        #pragma unroll
        for (int j = 0; j < 8; j++) { pf[j] = (_Float16)p1[j]; pf[8 + j] = (_Float16)p2[j]; }

        // rescale accumulators (alpha uniform across this lane's VGPRs) + P.V
        #pragma unroll
        for (int n = 0; n < 8; n++) {
            #pragma unroll
            for (int j = 0; j < 8; j++) o_acc[n][j] = o_acc[n][j] * alpha;
            v16h af = load_frag(Vbase + (size_t)(n * 16) * TP + k0, TP);  // V^T tile
            o_acc[n] = wmma16(af, pf, o_acc[n]);
        }

        // next tile's async copy must be complete, and all waves done with
        // this buffer (syncthreads fences outstanding LDS reads) before reuse
        wait_asynccnt0();
        __syncthreads();
    }

    // epilogue: O[q][vcol] = acc / l ; map to xo[b][h*16 + vcol/64][t][vcol%64]
    const int q = lane & 15;
    const int trow = qrow0 + q;
    if (trow < TMAX) {
        const float inv_l = 1.f / l_run;
        const int h = hb >> 3, b = hb & 7;
        #pragma unroll
        for (int n = 0; n < 8; n++) {
            int vb2 = wave * 128 + n * 16 + ((lane >> 4) << 3);
            #pragma unroll
            for (int j = 0; j < 8; j++) {
                int vcol = vb2 + j;
                int cv = vcol >> 6, ff = vcol & 63;
                xo[(((size_t)b * CC + h * CV + cv) * TMAX + trow) * FF + ff] =
                    o_acc[n][j] * inv_l;
            }
        }
    }
}

// ============================================================
// Kernel 3: final 1x1 conv + PReLU + CFN(over all C*F) + residual
// ============================================================
__global__ __launch_bounds__(256) void out_kernel(
    const float* __restrict__ xo, const float* __restrict__ x,
    const float* __restrict__ lw, const float* __restrict__ lb,
    const float* __restrict__ la, const float* __restrict__ lg,
    const float* __restrict__ lbe, float* __restrict__ out)
{
    const int t = blockIdx.x;
    const int b = blockIdx.y;
    const int tid = threadIdx.x;

    __shared__ float xs[CC * FF];
    __shared__ float wsm[CC * CC];
    __shared__ float gsum, gsq;

    for (int i = tid; i < CC * FF; i += 256) {
        int c = i >> 6, f = i & 63;
        xs[i]  = xo[(((size_t)b * CC + c) * TMAX + t) * FF + f];
        wsm[i] = lw[i];
    }
    if (tid == 0) { gsum = 0.f; gsq = 0.f; }
    __syncthreads();

    const int f = tid & 63;
    const int obase = tid >> 6;
    const float alpha = la[0];
    float acc[16];
    float ps = 0.f, pq = 0.f;
    #pragma unroll
    for (int k = 0; k < 16; k++) {
        int o = obase + 4 * k;
        float s = lb[o];
        for (int c = 0; c < CC; c++) s += wsm[o * CC + c] * xs[c * FF + f];
        s = s > 0.f ? s : alpha * s;
        acc[k] = s; ps += s; pq += s * s;
    }
    #pragma unroll
    for (int m = 16; m >= 1; m >>= 1) {
        ps += __shfl_xor(ps, m, 32);
        pq += __shfl_xor(pq, m, 32);
    }
    if ((tid & 31) == 0) { atomicAdd(&gsum, ps); atomicAdd(&gsq, pq); }
    __syncthreads();

    float mu   = gsum / 4096.f;
    float var  = gsq / 4096.f - mu * mu;
    float istd = 1.f / (sqrtf(fmaxf(var, 0.f)) + EPS);
    #pragma unroll
    for (int k = 0; k < 16; k++) {
        int o = obase + 4 * k;
        float y = (acc[k] - mu) * istd * lg[o * FF + f] + lbe[o * FF + f];
        size_t idx = (((size_t)b * CC + o) * TMAX + t) * FF + f;
        out[idx] = y + x[idx];
    }
}

// ============================================================
extern "C" void kernel_launch(void* const* d_in, const int* in_sizes, int n_in,
                              void* d_out, int out_size, void* d_ws, size_t ws_size,
                              hipStream_t stream) {
    (void)in_sizes; (void)n_in; (void)out_size; (void)ws_size;
    const float* x   = (const float*)d_in[0];
    const float* qw  = (const float*)d_in[1];
    const float* qb  = (const float*)d_in[2];
    const float* qa  = (const float*)d_in[3];
    const float* qg  = (const float*)d_in[4];
    const float* qbe = (const float*)d_in[5];
    const float* kw  = (const float*)d_in[6];
    const float* kb  = (const float*)d_in[7];
    const float* ka  = (const float*)d_in[8];
    const float* kg  = (const float*)d_in[9];
    const float* kbe = (const float*)d_in[10];
    const float* vw  = (const float*)d_in[11];
    const float* vb  = (const float*)d_in[12];
    const float* va  = (const float*)d_in[13];
    const float* vg  = (const float*)d_in[14];
    const float* vbe = (const float*)d_in[15];
    const float* lw  = (const float*)d_in[16];
    const float* lb  = (const float*)d_in[17];
    const float* la  = (const float*)d_in[18];
    const float* lg  = (const float*)d_in[19];
    const float* lbe = (const float*)d_in[20];

    // workspace: Qh 16MB | Kh 16MB | Vt 64MB | xo 125MB  (~222 MB total)
    char* ws = (char*)d_ws;
    const size_t qk_bytes = (size_t)HBN * TP * DQK * sizeof(_Float16);
    const size_t vt_bytes = (size_t)HBN * DV * TP * sizeof(_Float16);
    _Float16* Qh = (_Float16*)ws;
    _Float16* Kh = (_Float16*)(ws + qk_bytes);
    _Float16* Vt = (_Float16*)(ws + 2 * qk_bytes);
    float*    xo = (float*)   (ws + 2 * qk_bytes + vt_bytes);

    dim3 g1(TP, BB);
    qkv_kernel<<<g1, 256, 0, stream>>>(x, qw, qb, qa, qg, qbe,
                                       kw, kb, ka, kg, kbe,
                                       vw, vb, va, vg, vbe, Qh, Kh, Vt);

    dim3 g2((TMAX + 15) / 16, HBN);   // 63 x 32
    attn_kernel<<<g2, 256, 0, stream>>>(Qh, Kh, Vt, xo);

    dim3 g3(TMAX, BB);
    out_kernel<<<g3, 256, 0, stream>>>(xo, x, lw, lb, la, lg, lbe, (float*)d_out);
}
